// ContinuousTimeRNN_16844861735747
// MI455X (gfx1250) — matrix operand order repaired and checked by reading, not compiled, not run on hardware
//
#include <hip/hip_runtime.h>
#include <stdint.h>
#include <stddef.h>

// Problem constants (match reference)
#define B_DIM 4096
#define I_DIM 1024
#define H_DIM 2048
#define NSTEPS 10
#define DT_STEP 0.1f   // (T1-T0)/NUM_STEPS

typedef __attribute__((ext_vector_type(16))) __bf16 v16bf;
typedef __attribute__((ext_vector_type(8)))  float  v8f;

// ---------------------------------------------------------------------------
// fp32 -> bf16 round-to-nearest-even (bit trick; NaN edge cases irrelevant here)
// ---------------------------------------------------------------------------
static __device__ __forceinline__ unsigned short f32_to_bf16(float f) {
  unsigned int u = __float_as_uint(f);
  u += 0x7FFFu + ((u >> 16) & 1u);
  return (unsigned short)(u >> 16);
}

// Fast tanh: prefer gfx1250 hardware v_tanh_f32; else exp-based identity that
// saturates correctly for large |x| (exp overflow -> +inf -> rcp -> 0 -> 1).
static __device__ __forceinline__ float fast_tanh(float x) {
#if __has_builtin(__builtin_amdgcn_tanhf)
  return __builtin_amdgcn_tanhf(x);
#else
  float e = __expf(2.0f * x);
  return 1.0f - 2.0f * __builtin_amdgcn_rcpf(e + 1.0f);
#endif
}

// ---------------------------------------------------------------------------
// Elementwise converters
// ---------------------------------------------------------------------------
__global__ void cvt_f32_to_bf16_kernel(const float* __restrict__ src,
                                       unsigned short* __restrict__ dst, int n4) {
  int i = blockIdx.x * blockDim.x + threadIdx.x;
  if (i >= n4) return;
  float4 f = reinterpret_cast<const float4*>(src)[i];
  ushort4 o;
  o.x = f32_to_bf16(f.x); o.y = f32_to_bf16(f.y);
  o.z = f32_to_bf16(f.z); o.w = f32_to_bf16(f.w);
  reinterpret_cast<ushort4*>(dst)[i] = o;
}

// h0 -> (fp32 master state in d_out) + (bf16 copy for first GEMM)
__global__ void h_init_kernel(const float* __restrict__ h0,
                              float* __restrict__ h_f32,
                              unsigned short* __restrict__ h_bf16, int n4) {
  int i = blockIdx.x * blockDim.x + threadIdx.x;
  if (i >= n4) return;
  float4 f = reinterpret_cast<const float4*>(h0)[i];
  reinterpret_cast<float4*>(h_f32)[i] = f;
  ushort4 o;
  o.x = f32_to_bf16(f.x); o.y = f32_to_bf16(f.y);
  o.z = f32_to_bf16(f.z); o.w = f32_to_bf16(f.w);
  reinterpret_cast<ushort4*>(h_bf16)[i] = o;
}

// decay[n] = exp(-dt/tau[n]); store (decay, 1-decay) as float2 per column.
__global__ void decay_kernel(const float* __restrict__ tau,
                             float2* __restrict__ dcy, int n) {
  int i = blockIdx.x * blockDim.x + threadIdx.x;
  if (i >= n) return;
  float d = __expf(-DT_STEP / tau[i]);
  dcy[i] = make_float2(d, 1.0f - d);
}

// ---------------------------------------------------------------------------
// Fragment loads per CDNA5 WMMA VGPR layouts (CDNA5 ISA §7.12.2)
// A (16x32 bf16): lane m<16 holds K={0..7,16..23}; lane m+16 holds K={8..15,24..31}
// B (32x16 bf16): lane n<16 holds K={0..7,16..23}; lane n+16 holds K={8..15,24..31}
// ---------------------------------------------------------------------------
static __device__ __forceinline__ v16bf load_frag2(const unsigned short* p0,
                                                   const unsigned short* p1) {
  union { uint4 u[2]; v16bf v; } r;
  r.u[0] = *reinterpret_cast<const uint4*>(p0);
  r.u[1] = *reinterpret_cast<const uint4*>(p1);
  return r.v;
}

// ---------------------------------------------------------------------------
// Wave-level GEMM core: acc[4][4] (64x64 tile) += A[m0:+64, :K] * W[n0:+64, :K]^T
// A row-major [M,lda] bf16, W row-major [N,ldb] bf16 (so B matrix = W^T).
// ---------------------------------------------------------------------------
static __device__ __forceinline__ void wave_gemm_bf16(
    const unsigned short* __restrict__ A,
    const unsigned short* __restrict__ W,
    int lda, int ldb, int K, int m0, int n0, int lane, v8f acc[4][4]) {
  const int r  = lane & 15;
  const int hi = lane >> 4;
  const int kA = hi * 8;    // A: second lane-half starts at K=8 (and 24)
  const int kB = hi * 8;    // B: second lane-half starts at K=8 (and 24)

  const unsigned short* aRow[4];
  const unsigned short* bRow[4];
#pragma unroll
  for (int mi = 0; mi < 4; ++mi)
    aRow[mi] = A + (size_t)(m0 + mi * 16 + r) * (size_t)lda + kA;
#pragma unroll
  for (int ni = 0; ni < 4; ++ni)
    bRow[ni] = W + (size_t)(n0 + ni * 16 + r) * (size_t)ldb + kB;

  for (int k = 0; k < K; k += 32) {
    v16bf afr[4], bfr[4];
#pragma unroll
    for (int mi = 0; mi < 4; ++mi)
      afr[mi] = load_frag2(aRow[mi] + k, aRow[mi] + k + 16);
#pragma unroll
    for (int ni = 0; ni < 4; ++ni)
      bfr[ni] = load_frag2(bRow[ni] + k, bRow[ni] + k + 16);
#pragma unroll
    for (int mi = 0; mi < 4; ++mi)
#pragma unroll
      for (int ni = 0; ni < 4; ++ni)
        acc[mi][ni] = __builtin_amdgcn_wmma_f32_16x16x32_bf16(
            false, afr[mi], false, bfr[ni], (short)0, acc[mi][ni], false, false);
  }
}

// Block = 256 threads = 8 waves arranged 4(M) x 2(N); block tile 256x128.
#define WAVES_M 4
#define WAVES_N 2
#define BLK_M   (WAVES_M * 64)
#define BLK_N   (WAVES_N * 64)

// ---------------------------------------------------------------------------
// ih = x @ W_ih^T + b_ih   (M=B_DIM, N=H_DIM, K=I_DIM), fp32 output
// ---------------------------------------------------------------------------
__global__ __launch_bounds__(256, 1)
void gemm_ih_kernel(const unsigned short* __restrict__ xb,
                    const unsigned short* __restrict__ Wihb,
                    const float* __restrict__ b_ih,
                    float* __restrict__ ih) {
  const int lane = threadIdx.x & 31;
  const int wave = threadIdx.x >> 5;
  const int m0 = blockIdx.y * BLK_M + (wave & 3) * 64;
  const int n0 = blockIdx.x * BLK_N + (wave >> 2) * 64;

  v8f acc[4][4];
#pragma unroll
  for (int mi = 0; mi < 4; ++mi)
#pragma unroll
    for (int ni = 0; ni < 4; ++ni) acc[mi][ni] = (v8f){};

  wave_gemm_bf16(xb, Wihb, I_DIM, I_DIM, I_DIM, m0, n0, lane, acc);

  const int colOff  = lane & 15;
  const int rowHalf = (lane >> 4) * 8;
#pragma unroll
  for (int ni = 0; ni < 4; ++ni) {
    const int n = n0 + ni * 16 + colOff;
    const float bh = b_ih[n];
#pragma unroll
    for (int mi = 0; mi < 4; ++mi) {
#pragma unroll
      for (int v = 0; v < 8; ++v) {
        const int m = m0 + mi * 16 + rowHalf + v;
        ih[(size_t)m * H_DIM + n] = acc[mi][ni][v] + bh;
      }
    }
  }
}

// ---------------------------------------------------------------------------
// One recurrent step:
//   pre   = ih + h @ W_hh^T + b_hh
//   h_new = decay*h + (1-decay)*tanh(pre)
// Reads h (bf16 ping), writes h (fp32 master in-place per-element) + bf16 pong.
// ---------------------------------------------------------------------------
__global__ __launch_bounds__(256, 1)
void rnn_step_kernel(const unsigned short* __restrict__ hb_in,
                     const unsigned short* __restrict__ Whhb,
                     const float* __restrict__ ih,
                     const float* __restrict__ b_hh,
                     const float2* __restrict__ dcy_tab,
                     float* __restrict__ h_f32,
                     unsigned short* __restrict__ hb_out) {
  const int lane = threadIdx.x & 31;
  const int wave = threadIdx.x >> 5;
  const int m0 = blockIdx.y * BLK_M + (wave & 3) * 64;
  const int n0 = blockIdx.x * BLK_N + (wave >> 2) * 64;

  v8f acc[4][4];
#pragma unroll
  for (int mi = 0; mi < 4; ++mi)
#pragma unroll
    for (int ni = 0; ni < 4; ++ni) acc[mi][ni] = (v8f){};

  wave_gemm_bf16(hb_in, Whhb, H_DIM, H_DIM, H_DIM, m0, n0, lane, acc);

  const int colOff  = lane & 15;
  const int rowHalf = (lane >> 4) * 8;
#pragma unroll
  for (int ni = 0; ni < 4; ++ni) {
    const int n = n0 + ni * 16 + colOff;
    const float  bh = b_hh[n];
    const float2 dc = dcy_tab[n];   // (decay, 1-decay)
#pragma unroll
    for (int mi = 0; mi < 4; ++mi) {
#pragma unroll
      for (int v = 0; v < 8; ++v) {
        const int m = m0 + mi * 16 + rowHalf + v;
        const size_t idx = (size_t)m * H_DIM + n;
        const float pre = acc[mi][ni][v] + ih[idx] + bh;
        const float tgt = fast_tanh(pre);
        const float hnew = dc.x * h_f32[idx] + dc.y * tgt;
        h_f32[idx]  = hnew;
        hb_out[idx] = f32_to_bf16(hnew);
      }
    }
  }
}

// ---------------------------------------------------------------------------
// Host-side launcher
// ---------------------------------------------------------------------------
extern "C" void kernel_launch(void* const* d_in, const int* in_sizes, int n_in,
                              void* d_out, int out_size, void* d_ws, size_t ws_size,
                              hipStream_t stream) {
  (void)in_sizes; (void)n_in; (void)out_size; (void)ws_size;

  const float* x    = (const float*)d_in[0];   // [B, I]
  const float* h0   = (const float*)d_in[1];   // [B, H]
  const float* W_ih = (const float*)d_in[2];   // [H, I]
  const float* b_ih = (const float*)d_in[3];   // [H]
  const float* W_hh = (const float*)d_in[4];   // [H, H]
  const float* b_hh = (const float*)d_in[5];   // [H]
  const float* tau  = (const float*)d_in[6];   // [H]
  float* h_out = (float*)d_out;                // [B, H] fp32, doubles as h master

  uint8_t* ws = (uint8_t*)d_ws;
  // Workspace layout (bytes):
  unsigned short* xb   = (unsigned short*)(ws);                            // 8 MB
  unsigned short* Wihb = (unsigned short*)(ws + (size_t)8  * 1024 * 1024); // 4 MB
  unsigned short* Whhb = (unsigned short*)(ws + (size_t)12 * 1024 * 1024); // 8 MB
  float*          ih   = (float*)        (ws + (size_t)20 * 1024 * 1024); // 32 MB
  unsigned short* hb0  = (unsigned short*)(ws + (size_t)52 * 1024 * 1024); // 16 MB
  unsigned short* hb1  = (unsigned short*)(ws + (size_t)68 * 1024 * 1024); // 16 MB
  float2*         dcy  = (float2*)        (ws + (size_t)84 * 1024 * 1024); // 16 KB
  // total: ~84 MB

  const int threads = 256;

  // 1) Down-convert x, W_ih, W_hh to bf16; seed h state; decay table.
  {
    int n4 = (B_DIM * I_DIM) / 4;
    cvt_f32_to_bf16_kernel<<<(n4 + threads - 1) / threads, threads, 0, stream>>>(x, xb, n4);
  }
  {
    int n4 = (H_DIM * I_DIM) / 4;
    cvt_f32_to_bf16_kernel<<<(n4 + threads - 1) / threads, threads, 0, stream>>>(W_ih, Wihb, n4);
  }
  {
    int n4 = (H_DIM * H_DIM) / 4;
    cvt_f32_to_bf16_kernel<<<(n4 + threads - 1) / threads, threads, 0, stream>>>(W_hh, Whhb, n4);
  }
  {
    int n4 = (B_DIM * H_DIM) / 4;
    h_init_kernel<<<(n4 + threads - 1) / threads, threads, 0, stream>>>(h0, h_out, hb0, n4);
  }
  decay_kernel<<<(H_DIM + threads - 1) / threads, threads, 0, stream>>>(tau, dcy, H_DIM);

  // 2) ih = x @ W_ih^T + b_ih (computed once; time-invariant drive).
  dim3 grid(H_DIM / BLK_N, B_DIM / BLK_M);  // (16, 16)
  gemm_ih_kernel<<<grid, threads, 0, stream>>>(xb, Wihb, b_ih, ih);

  // 3) 10 recurrent steps, ping-ponging the bf16 h copy.
  for (int s = 0; s < NSTEPS; ++s) {
    const unsigned short* hin  = (s & 1) ? hb1 : hb0;
    unsigned short*       hout = (s & 1) ? hb0 : hb1;
    rnn_step_kernel<<<grid, threads, 0, stream>>>(hin, Whhb, ih, b_hh, dcy,
                                                  h_out, hout);
  }
}
